// BlindStatefulLSTM_68461778698433
// MI455X (gfx1250) — matrix operand order repaired and checked by reading, not compiled, not run on hardware
//
#include <hip/hip_runtime.h>
#include <hip/hip_bf16.h>
#include <math.h>

// ---------------- types ----------------
typedef __attribute__((ext_vector_type(16))) __bf16 v16bf;
typedef __attribute__((ext_vector_type(8)))  float  v8f;
typedef __attribute__((ext_vector_type(4)))  unsigned int u32x4;
typedef __attribute__((ext_vector_type(8)))  unsigned int u32x8;

struct alignas(32) BF16x16 { u32x4 lo, hi; };

__device__ __forceinline__ unsigned short f2bf(float f) {
  unsigned int u = __builtin_bit_cast(unsigned int, f);
  unsigned int r = u + 0x7FFFu + ((u >> 16) & 1u);
  return (unsigned short)(r >> 16);
}
__device__ __forceinline__ float bf2f(unsigned short h) {
  unsigned int u = ((unsigned int)h) << 16;
  return __builtin_bit_cast(float, u);
}
__device__ __forceinline__ float sigm(float x) { return 1.0f / (1.0f + __expf(-x)); }

__device__ __forceinline__ v16bf load_afrag(const unsigned short* p) {
  BF16x16 af;
  af.lo = *(const u32x4*)(p);
  af.hi = *(const u32x4*)(p + 8);
  return __builtin_bit_cast(v16bf, af);
}
__device__ __forceinline__ v16bf load_bfrag(const unsigned short* p) {
  u32x8 raw = *(const u32x8*)(p);
  return __builtin_bit_cast(v16bf, raw);
}

// ---------------- dims ----------------
#define LSEQ 64
#define EDIM 300
#define HDIM 300
#define HP   320          // padded hidden / embed
#define KP   640          // combined K  (x:320 | h:320)
#define NP   1280         // 4*HP gate rows
#define D2H  600          // 2H
#define NSEQ 2176         // 128 utterances + 2048 history
#define BROWS 16          // sequences per workgroup in LSTM
#define LTHREADS 128      // 4 waves
#define AK   1216         // attention padded dim (1200 -> 1216)

// =====================================================================
// prep: pad & convert LSTM weights into combined bf16 [1280 x 640]
// =====================================================================
__global__ void pad_lstm_w(const float* __restrict__ Wih,
                           const float* __restrict__ Whh,
                           unsigned short* __restrict__ Wc) {
  int n = blockIdx.x;                // 0..1279 padded gate row
  int gate = n / HP, idx = n % HP;
  int srcrow = gate * HDIM + idx;
  bool vrow = idx < HDIM;
  for (int k = threadIdx.x; k < KP; k += 256) {
    float v = 0.0f;
    if (vrow) {
      if (k < EDIM)                    v = Wih[(size_t)srcrow * EDIM + k];
      else if (k >= HP && k < HP+HDIM) v = Whh[(size_t)srcrow * HDIM + (k - HP)];
    }
    Wc[(size_t)n * KP + k] = f2bf(v);
  }
}

__global__ void pad_lstm_bias(const float* __restrict__ bih,
                              const float* __restrict__ bhh,
                              float* __restrict__ bp) {
  int n = blockIdx.x * 256 + threadIdx.x;
  if (n < NP) {
    int gate = n / HP, idx = n % HP;
    bp[n] = (idx < HDIM) ? (bih[gate*HDIM+idx] + bhh[gate*HDIM+idx]) : 0.0f;
  }
}

__global__ void pad_w1(const float* __restrict__ W1, unsigned short* __restrict__ W1p) {
  int r = blockIdx.x;                // 0..1215
  for (int k = threadIdx.x; k < AK; k += 256) {
    float v = (r < 1200 && k < 1200) ? W1[(size_t)r*1200 + k] : 0.0f;
    W1p[(size_t)r*AK + k] = f2bf(v);
  }
}

__global__ void pad_b1(const float* __restrict__ b1, float* __restrict__ b1p) {
  int n = blockIdx.x * 256 + threadIdx.x;
  if (n < AK) b1p[n] = (n < 1200) ? b1[n] : 0.0f;
}

// =====================================================================
// BiLSTM: one WG = 16 sequences (4 waves), all 64 timesteps, one direction
// gates[16,1280] = [x|h][16,640] @ Wc^T  via v_wmma_f32_16x16x32_bf16
// cell state + bias live in LDS (no dynamically-indexed register arrays)
// =====================================================================
__global__ void __launch_bounds__(LTHREADS)
lstm_kernel(const int* __restrict__ utt, const int* __restrict__ slen,
            const int* __restrict__ hist, const int* __restrict__ hlen,
            const float* __restrict__ emb,
            const unsigned short* __restrict__ Wf,
            const unsigned short* __restrict__ Wb,
            const float* __restrict__ biasf,
            const float* __restrict__ biasb,
            float* __restrict__ hout) {
  __shared__ __align__(16) unsigned short s_x [BROWS * HP];
  __shared__ __align__(16) unsigned short s_hA[BROWS * HP];
  __shared__ __align__(16) unsigned short s_hB[BROWS * HP];
  __shared__ float s_c[BROWS * HP];      // cell state (fp32)
  __shared__ float s_bias[NP];           // fused bias
  __shared__ int s_len[BROWS];
  __shared__ int s_tok[BROWS];

  const int tid  = threadIdx.x;
  const int dir  = blockIdx.y;
  const int base = blockIdx.x * BROWS;
  const unsigned short* W = dir ? Wb : Wf;
  const float* bias = dir ? biasb : biasf;

  if (tid < BROWS) {
    int s = base + tid;
    s_len[tid] = (s < 128) ? slen[s] : hlen[s - 128];
  }
  for (int i = tid; i < BROWS * HP; i += LTHREADS) {
    s_hA[i] = 0;                         // h0 = 0
    s_c[i]  = 0.0f;                      // c0 = 0
  }
  for (int i = tid; i < NP; i += LTHREADS) s_bias[i] = bias[i];
  __syncthreads();

  const int lane  = tid & 31;
  const int wave  = tid >> 5;           // 0..3
  const int jbase = wave * 5;           // 5 hidden tiles per wave
  const int lhalf = lane >> 4;
  const int lmod  = lane & 15;
  const int aoff  = lhalf ? 16 : 0;

  int len_e[8], row_e[8];
#pragma unroll
  for (int e = 0; e < 8; ++e) {
    row_e[e] = e + (lhalf ? 8 : 0);
    len_e[e] = s_len[row_e[e]];
  }

  unsigned short* hCur  = s_hA;
  unsigned short* hNext = s_hB;
  const unsigned short* apx = s_x + lmod * HP + aoff;   // A base in x region

  for (int t = 0; t < LSEQ; ++t) {
    if (tid < BROWS) {                  // token for this step (reversed prefix for bwd)
      int s = base + tid;
      int ts = dir ? (s_len[tid] - 1 - t) : t;
      if (ts < 0) ts = 0;
      s_tok[tid] = (s < 128) ? utt[s * LSEQ + ts] : hist[(size_t)(s - 128) * LSEQ + ts];
    }
    __syncthreads();
    for (int i = tid; i < BROWS * HP; i += LTHREADS) {   // gather embeddings -> bf16 LDS
      int r = i / HP, c = i - r * HP;
      float v = (c < EDIM) ? emb[(size_t)s_tok[r] * EDIM + c] : 0.0f;
      s_x[i] = f2bf(v);
    }
    __syncthreads();

    const unsigned short* aph = hCur + lmod * HP + aoff;  // A base in h region

    for (int jj = 0; jj < 5; ++jj) {
      const int jt = jbase + jj;
      // per-gate weight-row base pointers for this hidden tile
      const unsigned short* wb0 = W + (size_t)((0 * 20 + jt) * 16 + lmod) * KP + lhalf * 16;
      const unsigned short* wb1 = W + (size_t)((1 * 20 + jt) * 16 + lmod) * KP + lhalf * 16;
      const unsigned short* wb2 = W + (size_t)((2 * 20 + jt) * 16 + lmod) * KP + lhalf * 16;
      const unsigned short* wb3 = W + (size_t)((3 * 20 + jt) * 16 + lmod) * KP + lhalf * 16;
      v8f acc[4];
      // ---- kt = 0 peeled: C = inline zero ----
      {
        v16bf av = load_afrag(apx);
        const v8f zc = {};
        acc[0] = __builtin_amdgcn_wmma_f32_16x16x32_bf16(
            false, av, false, load_bfrag(wb0), (short)0, zc, false, false);
        acc[1] = __builtin_amdgcn_wmma_f32_16x16x32_bf16(
            false, av, false, load_bfrag(wb1), (short)0, zc, false, false);
        acc[2] = __builtin_amdgcn_wmma_f32_16x16x32_bf16(
            false, av, false, load_bfrag(wb2), (short)0, zc, false, false);
        acc[3] = __builtin_amdgcn_wmma_f32_16x16x32_bf16(
            false, av, false, load_bfrag(wb3), (short)0, zc, false, false);
      }
      // ---- x half: kt = 1..9 ----
      for (int kt = 1; kt < 10; ++kt) {
        v16bf av = load_afrag(apx + kt * 32);
        acc[0] = __builtin_amdgcn_wmma_f32_16x16x32_bf16(
            false, av, false, load_bfrag(wb0 + kt * 32), (short)0, acc[0], false, false);
        acc[1] = __builtin_amdgcn_wmma_f32_16x16x32_bf16(
            false, av, false, load_bfrag(wb1 + kt * 32), (short)0, acc[1], false, false);
        acc[2] = __builtin_amdgcn_wmma_f32_16x16x32_bf16(
            false, av, false, load_bfrag(wb2 + kt * 32), (short)0, acc[2], false, false);
        acc[3] = __builtin_amdgcn_wmma_f32_16x16x32_bf16(
            false, av, false, load_bfrag(wb3 + kt * 32), (short)0, acc[3], false, false);
      }
      // ---- h half: kt = 10..19 (A from hCur, W offset by 320 cols) ----
      for (int kk = 0; kk < 10; ++kk) {
        v16bf av = load_afrag(aph + kk * 32);
        acc[0] = __builtin_amdgcn_wmma_f32_16x16x32_bf16(
            false, av, false, load_bfrag(wb0 + HP + kk * 32), (short)0, acc[0], false, false);
        acc[1] = __builtin_amdgcn_wmma_f32_16x16x32_bf16(
            false, av, false, load_bfrag(wb1 + HP + kk * 32), (short)0, acc[1], false, false);
        acc[2] = __builtin_amdgcn_wmma_f32_16x16x32_bf16(
            false, av, false, load_bfrag(wb2 + HP + kk * 32), (short)0, acc[2], false, false);
        acc[3] = __builtin_amdgcn_wmma_f32_16x16x32_bf16(
            false, av, false, load_bfrag(wb3 + HP + kk * 32), (short)0, acc[3], false, false);
      }
      const int c0 = jt * 16 + lmod;
      const float bi = s_bias[(0 * 20 + jt) * 16 + lmod];
      const float bf_ = s_bias[(1 * 20 + jt) * 16 + lmod];
      const float bg = s_bias[(2 * 20 + jt) * 16 + lmod];
      const float bo = s_bias[(3 * 20 + jt) * 16 + lmod];
#pragma unroll
      for (int e = 0; e < 8; ++e) {
        int addr = row_e[e] * HP + c0;
        float iv = acc[0][e] + bi;
        float fv = acc[1][e] + bf_;
        float gv = acc[2][e] + bg;
        float ov = acc[3][e] + bo;
        float cp = s_c[addr];
        float cn = sigm(fv) * cp + sigm(iv) * tanhf(gv);
        float hn = sigm(ov) * tanhf(cn);
        bool live = (t < len_e[e]);
        s_c[addr]   = live ? cn : cp;
        hNext[addr] = live ? f2bf(hn) : hCur[addr];
      }
    }
    __syncthreads();
    unsigned short* tmp = hCur; hCur = hNext; hNext = tmp;
  }

  for (int i = tid; i < BROWS * HDIM; i += LTHREADS) {   // final hidden -> fp32
    int r = i / HDIM, c = i - r * HDIM;
    hout[(size_t)(base + r) * D2H + dir * HDIM + c] = bf2f(hCur[r * HP + c]);
  }
}

// =====================================================================
// layernorm in place over rows of h [2176, 600]
// =====================================================================
__global__ void ln_kernel(float* __restrict__ h, const float* __restrict__ g,
                          const float* __restrict__ b) {
  int r = blockIdx.x * 256 + threadIdx.x;
  if (r >= NSEQ) return;
  float* x = h + (size_t)r * D2H;
  float mu = 0.0f;
  for (int i = 0; i < D2H; ++i) mu += x[i];
  mu *= (1.0f / D2H);
  float var = 0.0f;
  for (int i = 0; i < D2H; ++i) { float d = x[i] - mu; var += d * d; }
  var *= (1.0f / D2H);
  float sc = rsqrtf(var + 1e-5f);
  for (int i = 0; i < D2H; ++i) x[i] = (x[i] - mu) * sc * g[i] + b[i];
}

// =====================================================================
// build attention input A[2048,1216] = bf16([u_t(b) | mem(b,t)] padded)
// =====================================================================
__global__ void build_attn_A(const float* __restrict__ h, unsigned short* __restrict__ A) {
  int r = blockIdx.x;                 // 0..2047  (= b*16 + t)
  int b = r >> 4;
  for (int c = threadIdx.x; c < AK; c += 256) {
    float v = 0.0f;
    if (c < D2H)            v = h[(size_t)b * D2H + c];
    else if (c < 2 * D2H)   v = h[(size_t)(128 + r) * D2H + (c - D2H)];
    A[(size_t)r * AK + c] = f2bf(v);
  }
}

// =====================================================================
// S = tanh(A @ W1^T + b1)   WMMA GEMM  M=2048 N=1216 K=1216
// =====================================================================
__global__ void attn_gemm_tanh(const unsigned short* __restrict__ A,
                               const unsigned short* __restrict__ Bw,
                               const float* __restrict__ bias,
                               unsigned short* __restrict__ S) {
  const int tid = threadIdx.x;
  const int lane = tid & 31, wave = tid >> 5;
  const int mt = wave >> 2, nt = wave & 3;
  const int lmod = lane & 15, lhalf = lane >> 4;
  const int Mb = blockIdx.x * 32 + mt * 16;
  const int Nb = blockIdx.y * 64 + nt * 16;
  const unsigned short* ap0 = A  + (size_t)(Mb + lmod) * AK + (lhalf ? 16 : 0);
  const unsigned short* bp0 = Bw + (size_t)(Nb + lmod) * AK + lhalf * 16;
  v8f acc;
  {
    const v8f zc = {};
    acc = __builtin_amdgcn_wmma_f32_16x16x32_bf16(
        false, load_afrag(ap0), false, load_bfrag(bp0), (short)0, zc, false, false);
  }
  for (int kt = 1; kt < AK / 32; ++kt) {
    acc = __builtin_amdgcn_wmma_f32_16x16x32_bf16(
        false, load_afrag(ap0 + kt * 32), false, load_bfrag(bp0 + kt * 32),
        (short)0, acc, false, false);
  }
  float bv = bias[Nb + lmod];
#pragma unroll
  for (int e = 0; e < 8; ++e) {
    int row = Mb + e + (lhalf ? 8 : 0);
    S[(size_t)row * AK + Nb + lmod] = f2bf(tanhf(acc[e] + bv));
  }
}

// score[r] = S[r,:1200] . W2 + b2
__global__ void score_kernel(const unsigned short* __restrict__ S,
                             const float* __restrict__ W2, const float* __restrict__ b2,
                             float* __restrict__ sc) {
  int r = blockIdx.x * 256 + threadIdx.x;
  if (r >= 2048) return;
  float s = b2[0];
  const unsigned short* row = S + (size_t)r * AK;
  for (int k = 0; k < 1200; ++k) s += bf2f(row[k]) * W2[k];
  sc[r] = s;
}

// softmax over valid turns + weighted sum of memory -> c_t [128,600]
__global__ void softmax_ctx(const float* __restrict__ sc, const int* __restrict__ nturn,
                            const float* __restrict__ h, float* __restrict__ ct) {
  __shared__ float sw[16];
  int b = blockIdx.x, tid = threadIdx.x;
  if (tid < 16) sw[tid] = (tid < nturn[b]) ? sc[b * 16 + tid] : -1e9f;
  __syncthreads();
  if (tid == 0) {
    float mx = sw[0];
    for (int t = 1; t < 16; ++t) mx = fmaxf(mx, sw[t]);
    float se = 0.0f;
    for (int t = 0; t < 16; ++t) { sw[t] = __expf(sw[t] - mx); se += sw[t]; }
    float inv = 1.0f / se;
    for (int t = 0; t < 16; ++t) sw[t] *= inv;
  }
  __syncthreads();
  for (int i = tid; i < D2H; i += 256) {
    float a = 0.0f;
    for (int t = 0; t < 16; ++t)
      a += sw[t] * h[(size_t)(128 + b * 16 + t) * D2H + i];
    ct[(size_t)b * D2H + i] = a;
  }
}

// =====================================================================
// heads: c1 = relu(Wap@[u;ct]+b), c2 = relu(Watp@[u;c1]+b), logits + probs
// =====================================================================
__global__ void head_kernel(const float* __restrict__ h, const float* __restrict__ ct,
                            const float* __restrict__ Wap, const float* __restrict__ bap,
                            const float* __restrict__ Watp, const float* __restrict__ batp,
                            const float* __restrict__ Wao, const float* __restrict__ bao,
                            const float* __restrict__ Wro, const float* __restrict__ bro,
                            float* __restrict__ out) {
  __shared__ float s_cat[2 * D2H];
  __shared__ float s_c1[D2H];
  __shared__ float s_c2[D2H];
  __shared__ float s_log[16];
  int b = blockIdx.x, tid = threadIdx.x;
  for (int i = tid; i < D2H; i += 256) {
    s_cat[i]       = h[(size_t)b * D2H + i];
    s_cat[D2H + i] = ct[(size_t)b * D2H + i];
  }
  __syncthreads();
  for (int o = tid; o < D2H; o += 256) {
    float s = bap[o];
    const float* w = Wap + (size_t)o * (2 * D2H);
    for (int k = 0; k < 2 * D2H; ++k) s += s_cat[k] * w[k];
    s_c1[o] = fmaxf(s, 0.0f);
  }
  __syncthreads();
  for (int i = tid; i < D2H; i += 256) s_cat[D2H + i] = s_c1[i];
  __syncthreads();
  for (int o = tid; o < D2H; o += 256) {
    float s = batp[o];
    const float* w = Watp + (size_t)o * (2 * D2H);
    for (int k = 0; k < 2 * D2H; ++k) s += s_cat[k] * w[k];
    s_c2[o] = fmaxf(s, 0.0f);
  }
  __syncthreads();
  float* act_out   = out;
  float* attrs_out = out + 128 * 16;
  float* act_p     = out + 128 * (16 + 33);
  float* attrs_p   = out + 128 * (16 + 33 + 16);
  if (tid < 16) {
    float s = bao[tid];
    const float* w = Wao + (size_t)tid * D2H;
    for (int k = 0; k < D2H; ++k) s += s_c1[k] * w[k];
    s_log[tid] = s;
    act_out[b * 16 + tid] = s;
  } else if (tid < 49) {
    int o = tid - 16;
    float s = bro[o];
    const float* w = Wro + (size_t)o * D2H;
    for (int k = 0; k < D2H; ++k) s += s_c2[k] * w[k];
    attrs_out[b * 33 + o] = s;
    attrs_p[b * 33 + o]   = 1.0f / (1.0f + __expf(-s));
  }
  __syncthreads();
  if (tid == 0) {
    float mx = s_log[0];
    for (int o = 1; o < 16; ++o) mx = fmaxf(mx, s_log[o]);
    float se = 0.0f;
    float e[16];
    for (int o = 0; o < 16; ++o) { e[o] = __expf(s_log[o] - mx); se += e[o]; }
    float inv = 1.0f / se;
    for (int o = 0; o < 16; ++o) act_p[b * 16 + o] = e[o] * inv;
  }
}

// =====================================================================
// host launcher
// =====================================================================
extern "C" void kernel_launch(void* const* d_in, const int* in_sizes, int n_in,
                              void* d_out, int out_size, void* d_ws, size_t ws_size,
                              hipStream_t stream) {
  const int*   utterances = (const int*)  d_in[0];
  const int*   seq_len    = (const int*)  d_in[1];
  const int*   hist_tok   = (const int*)  d_in[2];
  const int*   hist_len   = (const int*)  d_in[3];
  const int*   num_turns  = (const int*)  d_in[4];
  const float* emb        = (const float*)d_in[5];
  const float* Wih_f      = (const float*)d_in[6];
  const float* Whh_f      = (const float*)d_in[7];
  const float* bih_f      = (const float*)d_in[8];
  const float* bhh_f      = (const float*)d_in[9];
  const float* Wih_b      = (const float*)d_in[10];
  const float* Whh_b      = (const float*)d_in[11];
  const float* bih_b      = (const float*)d_in[12];
  const float* bhh_b      = (const float*)d_in[13];
  const float* ln_g       = (const float*)d_in[14];
  const float* ln_b       = (const float*)d_in[15];
  const float* attn_W1    = (const float*)d_in[16];
  const float* attn_b1    = (const float*)d_in[17];
  const float* attn_W2    = (const float*)d_in[18];
  const float* attn_b2    = (const float*)d_in[19];
  const float* W_act_post = (const float*)d_in[20];
  const float* b_act_post = (const float*)d_in[21];
  const float* W_attr_post= (const float*)d_in[22];
  const float* b_attr_post= (const float*)d_in[23];
  const float* W_act_out  = (const float*)d_in[24];
  const float* b_act_out  = (const float*)d_in[25];
  const float* W_attr_out = (const float*)d_in[26];
  const float* b_attr_out = (const float*)d_in[27];

  char* ws = (char*)d_ws;
  size_t off = 0;
  auto take = [&](size_t bytes) { size_t o = off; off += (bytes + 255) & ~(size_t)255; return o; };
  unsigned short* Wc_f  = (unsigned short*)(ws + take((size_t)NP * KP * 2));
  unsigned short* Wc_b  = (unsigned short*)(ws + take((size_t)NP * KP * 2));
  float*          bp_f  = (float*)         (ws + take((size_t)NP * 4));
  float*          bp_b  = (float*)         (ws + take((size_t)NP * 4));
  float*          hfin  = (float*)         (ws + take((size_t)NSEQ * D2H * 4));
  unsigned short* Apad  = (unsigned short*)(ws + take((size_t)2048 * AK * 2));
  unsigned short* W1p   = (unsigned short*)(ws + take((size_t)AK * AK * 2));
  float*          b1p   = (float*)         (ws + take((size_t)AK * 4));
  unsigned short* S1    = (unsigned short*)(ws + take((size_t)2048 * AK * 2));
  float*          sc    = (float*)         (ws + take((size_t)2048 * 4));
  float*          ctb   = (float*)         (ws + take((size_t)128 * D2H * 4));
  (void)ws_size; (void)n_in; (void)in_sizes; (void)out_size;

  // weight prep
  pad_lstm_w   <<<NP, 256, 0, stream>>>(Wih_f, Whh_f, Wc_f);
  pad_lstm_w   <<<NP, 256, 0, stream>>>(Wih_b, Whh_b, Wc_b);
  pad_lstm_bias<<<(NP + 255) / 256, 256, 0, stream>>>(bih_f, bhh_f, bp_f);
  pad_lstm_bias<<<(NP + 255) / 256, 256, 0, stream>>>(bih_b, bhh_b, bp_b);
  pad_w1       <<<AK, 256, 0, stream>>>(attn_W1, W1p);
  pad_b1       <<<(AK + 255) / 256, 256, 0, stream>>>(attn_b1, b1p);

  // BiLSTM over utterances + history (both directions concurrently)
  lstm_kernel<<<dim3(NSEQ / BROWS, 2), LTHREADS, 0, stream>>>(
      utterances, seq_len, hist_tok, hist_len, emb,
      Wc_f, Wc_b, bp_f, bp_b, hfin);

  // layernorm (in place)
  ln_kernel<<<(NSEQ + 255) / 256, 256, 0, stream>>>(hfin, ln_g, ln_b);

  // attention
  build_attn_A<<<2048, 256, 0, stream>>>(hfin, Apad);
  attn_gemm_tanh<<<dim3(2048 / 32, AK / 64), 256, 0, stream>>>(Apad, W1p, b1p, S1);
  score_kernel<<<2048 / 256, 256, 0, stream>>>(S1, attn_W2, attn_b2, sc);
  softmax_ctx<<<128, 256, 0, stream>>>(sc, num_turns, hfin, ctb);

  // heads
  head_kernel<<<128, 256, 0, stream>>>(hfin, ctb,
      W_act_post, b_act_post, W_attr_post, b_attr_post,
      W_act_out, b_act_out, W_attr_out, b_attr_out, (float*)d_out);
}
